// LinearBlock_49709951484049
// MI455X (gfx1250) — matrix operand order repaired and checked by reading, not compiled, not run on hardware
//
#include <hip/hip_runtime.h>
#include <math.h>

// ---------------------------------------------------------------------------
// Problem constants (reference: B=2, T=1024, D=1024, H=16, DH=64, FF=4096)
// ---------------------------------------------------------------------------
#define BB   2
#define TT   1024
#define DD   1024
#define HH   16
#define DH   64
#define FF_  4096
#define MROWS (BB * TT)   // 2048
#define MM   (1024 * 1024)

typedef __bf16 bf16;
typedef __attribute__((ext_vector_type(16))) __bf16 v16bf;
typedef __attribute__((ext_vector_type(8)))  __bf16 v8bf;
typedef __attribute__((ext_vector_type(4)))  __bf16 v4bf;
typedef __attribute__((ext_vector_type(8)))  float  v8f;

__device__ __forceinline__ bf16 f2bf(float f) {
    // round-to-nearest-even f32 -> bf16
    unsigned u = __builtin_bit_cast(unsigned, f);
    u += 0x7FFFu + ((u >> 16) & 1u);
    unsigned short s = (unsigned short)(u >> 16);
    return __builtin_bit_cast(bf16, s);
}
__device__ __forceinline__ float bf2f(bf16 x) {
    unsigned short s = __builtin_bit_cast(unsigned short, x);
    unsigned u = ((unsigned)s) << 16;
    return __builtin_bit_cast(float, u);
}

// ---------------------------------------------------------------------------
// Weight transpose + f32->bf16 convert:  W[K][N] -> Wt[N][K]  (K-major bf16)
// ---------------------------------------------------------------------------
__global__ __launch_bounds__(256) void wconv_t(const float* __restrict__ W,
                                               bf16* __restrict__ Wt,
                                               int K, int N) {
    __shared__ bf16 tile[32][33];
    const int k0 = blockIdx.y * 32;
    const int n0 = blockIdx.x * 32;
    const int e = threadIdx.x * 4;
    const int r = e >> 5;    // 0..31
    const int c = e & 31;    // 0,4,...,28

    float4 f = *(const float4*)(W + (size_t)(k0 + r) * N + n0 + c);
    tile[r][c + 0] = f2bf(f.x);
    tile[r][c + 1] = f2bf(f.y);
    tile[r][c + 2] = f2bf(f.z);
    tile[r][c + 3] = f2bf(f.w);
    __syncthreads();

    bf16* dst = Wt + (size_t)(n0 + r) * K + k0 + c;
    dst[0] = tile[c + 0][r];
    dst[1] = tile[c + 1][r];
    dst[2] = tile[c + 2][r];
    dst[3] = tile[c + 3][r];
}

// ---------------------------------------------------------------------------
// LayerNorm: one block per row (D = 1024, 256 threads x 4), bf16 output
// ---------------------------------------------------------------------------
__global__ __launch_bounds__(256) void ln_kernel(const float* __restrict__ x,
                                                 const float* __restrict__ g,
                                                 const float* __restrict__ b,
                                                 bf16* __restrict__ y) {
    const int row = blockIdx.x;
    float4 v = ((const float4*)(x + (size_t)row * DD))[threadIdx.x];

    __shared__ float red[256];
    red[threadIdx.x] = v.x + v.y + v.z + v.w;
    __syncthreads();
    for (int off = 128; off > 0; off >>= 1) {
        if (threadIdx.x < off) red[threadIdx.x] += red[threadIdx.x + off];
        __syncthreads();
    }
    const float mean = red[0] * (1.0f / DD);
    __syncthreads();

    float dx = v.x - mean, dy = v.y - mean, dz = v.z - mean, dw = v.w - mean;
    red[threadIdx.x] = dx * dx + dy * dy + dz * dz + dw * dw;
    __syncthreads();
    for (int off = 128; off > 0; off >>= 1) {
        if (threadIdx.x < off) red[threadIdx.x] += red[threadIdx.x + off];
        __syncthreads();
    }
    const float rstd = rsqrtf(red[0] * (1.0f / DD) + 1e-5f);

    float4 gg = ((const float4*)g)[threadIdx.x];
    float4 bb = ((const float4*)b)[threadIdx.x];
    v4bf o;
    o[0] = f2bf(dx * rstd * gg.x + bb.x);
    o[1] = f2bf(dy * rstd * gg.y + bb.y);
    o[2] = f2bf(dz * rstd * gg.z + bb.z);
    o[3] = f2bf(dw * rstd * gg.w + bb.w);
    ((v4bf*)(y + (size_t)row * DD))[threadIdx.x] = o;
}

// ---------------------------------------------------------------------------
// WMMA bf16 GEMM:  C[M,N] = A[M,K] @ B[K,N]  (+bias) (gelu) (+res f32)
//   A: bf16 row-major [M][K];  Bt: bf16 K-major [N][K]  (pre-transposed)
// Block: 256 threads (8 waves), 128x64 output tile, BK = 64.
// 8 waves in a 4(M) x 2(N) grid; each wave owns a 2x2 grid of 16x16 tiles
// -> per K-32 step: 8 ds_load_b128 feed 4 v_wmma (A and B fragments reused).
// Tiles stream into DOUBLE-BUFFERED LDS via global_load_async_to_lds_b128
// (ASYNCcnt DMA, no VGPR round-trip); tile i+1 copy overlaps tile i WMMAs;
// one barrier per K iteration.
//
// Fragment layouts per CDNA5 ISA 7.12.2 (16-bit, wave32):
//   A: lane m = L%16; half = (L<16?0:8); VGPR0..3 K=half..half+7,
//      VGPR4..7 K=16+half..16+half+7
//   B: lane n = L%16; half = (L<16?0:8); same K order as A
//   C/D f32: VGPR r: lanes0-15 (M=r,N=lane); lanes16-31 (M=r+8,N=lane-16)
// ---------------------------------------------------------------------------
__global__ __launch_bounds__(256) void gemm_wmma_bf16(
        const bf16* __restrict__ A, const bf16* __restrict__ Bt,
        const float* __restrict__ bias, const float* __restrict__ res,
        float* __restrict__ Cf, bf16* __restrict__ Cb,
        int M, int N, int K, int do_gelu) {
    __shared__ __align__(16) bf16 As[2][128 * 64];  // [m][k]  16 KB per buf
    __shared__ __align__(16) bf16 Bs[2][64 * 64];   // [n][k]   8 KB per buf

    const int tid  = threadIdx.x;
    const int lane = tid & 31;
    const int wv   = tid >> 5;
    const int wr   = wv >> 1;       // 0..3  (M direction)
    const int wc   = wv & 1;        // 0..1  (N direction)

    const int m0 = blockIdx.y * 128;
    const int n0 = blockIdx.x * 64;

    // cooperative-copy mapping
    // A tile: 128x64 bf16 = 8192 elems / 256 thr = 32 each (64 B, 4 x b128)
    const int a_row = tid >> 1;          // 0..127
    const int a_col = (tid & 1) * 32;    // 0,32
    // B tile: 64x64 bf16 = 4096 elems / 256 thr = 16 each (32 B, 2 x b128)
    const int b_row = tid >> 2;          // 0..63
    const int b_col = (tid & 3) * 16;    // 0,16,32,48

    const int mfrag = lane & 15;
    const int ahalf = (lane < 16) ? 0 : 8;
    const int bkoff = (lane < 16) ? 0 : 16;

    const int ti0 = wr * 2;             // wave's two M sub-tiles
    const int ti1 = wr * 2 + 1;
    const int tj0 = wc * 2;             // wave's two N sub-tiles
    const int tj1 = wc * 2 + 1;

    // per-thread source pointers (advance by kt)
    const bf16* gA = A  + (size_t)(m0 + a_row) * K + a_col;
    const bf16* gB = Bt + (size_t)(n0 + b_row) * K + b_col;

    v8f acc00 = {0.f, 0.f, 0.f, 0.f, 0.f, 0.f, 0.f, 0.f};
    v8f acc01 = acc00, acc10 = acc00, acc11 = acc00;

    // Issue async DMA of one K-slice (A 128x64 + B 64x64) into buffer `buf`.
    // inst offset applies to BOTH LDS and global address (ISA 15.18.3).
    auto issue_tile = [&](int buf, int kt) {
        unsigned la = (unsigned)(size_t)&As[buf][a_row * 64 + a_col];
        unsigned lb = (unsigned)(size_t)&Bs[buf][b_row * 64 + b_col];
        unsigned long long ga = (unsigned long long)(size_t)(gA + kt);
        unsigned long long gb = (unsigned long long)(size_t)(gB + kt);
        asm volatile(
            "global_load_async_to_lds_b128 %0, %2, off\n\t"
            "global_load_async_to_lds_b128 %0, %2, off offset:16\n\t"
            "global_load_async_to_lds_b128 %0, %2, off offset:32\n\t"
            "global_load_async_to_lds_b128 %0, %2, off offset:48\n\t"
            "global_load_async_to_lds_b128 %1, %3, off\n\t"
            "global_load_async_to_lds_b128 %1, %3, off offset:16"
            :: "v"(la), "v"(lb), "v"(ga), "v"(gb)
            : "memory");
    };

    const int nk = K >> 6;
    issue_tile(0, 0);
    int cur = 0;

    for (int it = 0; it < nk; ++it) {
        // make tile `it` visible (own DMA drained, then cross-wave barrier)
        asm volatile("s_wait_asynccnt 0x0" ::: "memory");
        __syncthreads();
        // overlap: start DMA of the next tile into the other buffer
        if (it + 1 < nk) issue_tile(cur ^ 1, (it + 1) * 64);

        const bf16* as = As[cur];
        const bf16* bs = Bs[cur];
#pragma unroll
        for (int kk = 0; kk < 64; kk += 32) {
            v8bf t0, t1;
            t0 = *(const v8bf*)(as + (ti0 * 16 + mfrag) * 64 + kk + ahalf);
            t1 = *(const v8bf*)(as + (ti0 * 16 + mfrag) * 64 + kk + 16 + ahalf);
            v16bf a0 = __builtin_shufflevector(t0, t1,
                        0,1,2,3,4,5,6,7,8,9,10,11,12,13,14,15);
            t0 = *(const v8bf*)(as + (ti1 * 16 + mfrag) * 64 + kk + ahalf);
            t1 = *(const v8bf*)(as + (ti1 * 16 + mfrag) * 64 + kk + 16 + ahalf);
            v16bf a1 = __builtin_shufflevector(t0, t1,
                        0,1,2,3,4,5,6,7,8,9,10,11,12,13,14,15);
            t0 = *(const v8bf*)(bs + (tj0 * 16 + mfrag) * 64 + kk + ahalf);
            t1 = *(const v8bf*)(bs + (tj0 * 16 + mfrag) * 64 + kk + 16 + ahalf);
            v16bf b0 = __builtin_shufflevector(t0, t1,
                        0,1,2,3,4,5,6,7,8,9,10,11,12,13,14,15);
            t0 = *(const v8bf*)(bs + (tj1 * 16 + mfrag) * 64 + kk + ahalf);
            t1 = *(const v8bf*)(bs + (tj1 * 16 + mfrag) * 64 + kk + 16 + ahalf);
            v16bf b1 = __builtin_shufflevector(t0, t1,
                        0,1,2,3,4,5,6,7,8,9,10,11,12,13,14,15);

            acc00 = __builtin_amdgcn_wmma_f32_16x16x32_bf16(
                        false, a0, false, b0, (short)0, acc00, false, false);
            acc01 = __builtin_amdgcn_wmma_f32_16x16x32_bf16(
                        false, a0, false, b1, (short)0, acc01, false, false);
            acc10 = __builtin_amdgcn_wmma_f32_16x16x32_bf16(
                        false, a1, false, b0, (short)0, acc10, false, false);
            acc11 = __builtin_amdgcn_wmma_f32_16x16x32_bf16(
                        false, a1, false, b1, (short)0, acc11, false, false);
        }
        cur ^= 1;
    }

    // epilogue: 4 sub-tiles per wave
    const int rowoff = (lane < 16) ? 0 : 8;
    const int gcol0 = n0 + tj0 * 16 + mfrag;
    const int gcol1 = n0 + tj1 * 16 + mfrag;
    const float bv0 = bias ? bias[gcol0] : 0.0f;
    const float bv1 = bias ? bias[gcol1] : 0.0f;

    auto emit = [&](const v8f& acc, int ti, int gcol, float bval) {
#pragma unroll
        for (int r = 0; r < 8; ++r) {
            const int grow = m0 + ti * 16 + r + rowoff;
            float vv = acc[r] + bval;
            if (do_gelu) vv = 0.5f * vv * (1.0f + erff(vv * 0.70710678118654752f));
            if (res) vv += res[(size_t)grow * N + gcol];
            if (Cb) Cb[(size_t)grow * N + gcol] = f2bf(vv);
            else    Cf[(size_t)grow * N + gcol] = vv;
        }
    };
    emit(acc00, ti0, gcol0, bv0);
    emit(acc01, ti0, gcol1, bv1);
    emit(acc10, ti1, gcol0, bv0);
    emit(acc11, ti1, gcol1, bv1);
}

// ---------------------------------------------------------------------------
// Causal linear-attention scan over fused bf16 QKV buffer [M][3072]
// (cols 0-1023 = q, 1024-2047 = k, 2048-3071 = v). One block per (b,h);
// thread d owns column d of the 64x64 running state S. phi(x) = elu(x)+1.
// ---------------------------------------------------------------------------
__device__ __forceinline__ float phi_fm(float x) {
    return x > 0.0f ? x + 1.0f : __expf(x);
}

__global__ __launch_bounds__(64) void linattn_scan(
        const bf16* __restrict__ qkv, const int* __restrict__ mask,
        bf16* __restrict__ out) {
    const int bh = blockIdx.x;       // 0..B*H-1
    const int b  = bh >> 4;          // H = 16
    const int h  = bh & 15;
    const int d  = threadIdx.x;      // 0..63

    __shared__ __align__(16) float qs[64];
    __shared__ __align__(16) float ks[64];
    __shared__ __align__(16) float zs[64];

    float S[64];
#pragma unroll
    for (int f = 0; f < 64; ++f) S[f] = 0.0f;
    zs[d] = 0.0f;
    __syncthreads();

    for (int t = 0; t < TT; ++t) {
        const size_t rb = ((size_t)(b * TT + t)) * 3072 + h * DH;
        const float m = (float)mask[b * TT + t];
        const float kf = phi_fm(bf2f(qkv[rb + 1024 + d])) * m;
        qs[d] = phi_fm(bf2f(qkv[rb + d]));
        ks[d] = kf;
        zs[d] += kf;
        __syncthreads();

        const float vd = bf2f(qkv[rb + 2048 + d]) * m;
        float num = 0.0f, den = 0.0f;
#pragma unroll
        for (int j = 0; j < 16; ++j) {
            float4 k4 = ((const float4*)ks)[j];
            float4 q4 = ((const float4*)qs)[j];
            float4 z4 = ((const float4*)zs)[j];
            S[4 * j + 0] += k4.x * vd; num += q4.x * S[4 * j + 0]; den += q4.x * z4.x;
            S[4 * j + 1] += k4.y * vd; num += q4.y * S[4 * j + 1]; den += q4.y * z4.y;
            S[4 * j + 2] += k4.z * vd; num += q4.z * S[4 * j + 2]; den += q4.z * z4.z;
            S[4 * j + 3] += k4.w * vd; num += q4.w * S[4 * j + 3]; den += q4.w * z4.w;
        }
        out[((size_t)(b * TT + t)) * DD + h * DH + d] = f2bf(num / (den + 1e-6f));
        __syncthreads();
    }
}

// ---------------------------------------------------------------------------
// Orchestration.  Workspace (bf16 elems unless noted):
//   Wqkvt 3M | Wot 1M | W1t 4M | W2t 4M | h1b 2M | qkvb 6M | attnb 2M
//   h2b 2M | ffb 8M | x2 (f32) 2M           total ~72 MB
// ---------------------------------------------------------------------------
extern "C" void kernel_launch(void* const* d_in, const int* in_sizes, int n_in,
                              void* d_out, int out_size, void* d_ws, size_t ws_size,
                              hipStream_t stream) {
    (void)in_sizes; (void)n_in; (void)out_size; (void)ws_size;

    const float* x        = (const float*)d_in[0];
    const int*   attnmask = (const int*)  d_in[1];
    const float* Wq       = (const float*)d_in[2];
    const float* Wk       = (const float*)d_in[3];
    const float* Wv       = (const float*)d_in[4];
    const float* Wo       = (const float*)d_in[5];
    const float* ln_a_g   = (const float*)d_in[6];
    const float* ln_a_b   = (const float*)d_in[7];
    const float* ln_m_g   = (const float*)d_in[8];
    const float* ln_m_b   = (const float*)d_in[9];
    const float* W1       = (const float*)d_in[10];
    const float* b1       = (const float*)d_in[11];
    const float* W2       = (const float*)d_in[12];
    const float* b2       = (const float*)d_in[13];

    bf16* wsb = (bf16*)d_ws;
    size_t off = 0;
    bf16* Wqkvt = wsb + off; off += (size_t)3 * MM;   // [3072][1024]
    bf16* Wot   = wsb + off; off += (size_t)1 * MM;   // [1024][1024]
    bf16* W1t   = wsb + off; off += (size_t)4 * MM;   // [4096][1024]
    bf16* W2t   = wsb + off; off += (size_t)4 * MM;   // [1024][4096]
    bf16* h1b   = wsb + off; off += (size_t)2 * MM;   // [2048][1024]
    bf16* qkvb  = wsb + off; off += (size_t)6 * MM;   // [2048][3072]
    bf16* attnb = wsb + off; off += (size_t)2 * MM;   // [2048][1024]
    bf16* h2b   = wsb + off; off += (size_t)2 * MM;   // [2048][1024]
    bf16* ffb   = wsb + off; off += (size_t)8 * MM;   // [2048][4096]
    float* x2   = (float*)(wsb + off);                // [2048][1024] f32

    const dim3 blk(256);

    // 0) weight transpose + convert (bf16, K-major); Wq/Wk/Wv fused
    wconv_t<<<dim3(32, 32),  blk, 0, stream>>>(Wq, Wqkvt + (size_t)0 * MM, DD, DD);
    wconv_t<<<dim3(32, 32),  blk, 0, stream>>>(Wk, Wqkvt + (size_t)1 * MM, DD, DD);
    wconv_t<<<dim3(32, 32),  blk, 0, stream>>>(Wv, Wqkvt + (size_t)2 * MM, DD, DD);
    wconv_t<<<dim3(32, 32),  blk, 0, stream>>>(Wo, Wot, DD, DD);
    wconv_t<<<dim3(128, 32), blk, 0, stream>>>(W1, W1t, DD, FF_);
    wconv_t<<<dim3(32, 128), blk, 0, stream>>>(W2, W2t, FF_, DD);

    // 1) LN1
    ln_kernel<<<MROWS, blk, 0, stream>>>(x, ln_a_g, ln_a_b, h1b);
    // 2) fused QKV projection (N = 3072)
    gemm_wmma_bf16<<<dim3(48, 16), blk, 0, stream>>>(
        h1b, Wqkvt, nullptr, nullptr, nullptr, qkvb, MROWS, 3 * DD, DD, 0);
    // 3) causal linear-attention scan
    linattn_scan<<<BB * HH, 64, 0, stream>>>(qkvb, attnmask, attnb);
    // 4) output projection + residual -> x2 (f32)
    gemm_wmma_bf16<<<dim3(16, 16), blk, 0, stream>>>(
        attnb, Wot, nullptr, x, x2, nullptr, MROWS, DD, DD, 0);
    // 5) LN2
    ln_kernel<<<MROWS, blk, 0, stream>>>(x2, ln_m_g, ln_m_b, h2b);
    // 6) MLP up + GELU -> ffb (bf16)
    gemm_wmma_bf16<<<dim3(64, 16), blk, 0, stream>>>(
        h2b, W1t, b1, nullptr, nullptr, ffb, MROWS, FF_, DD, 1);
    // 7) MLP down + bias + residual -> d_out (f32)
    gemm_wmma_bf16<<<dim3(16, 16), blk, 0, stream>>>(
        ffb, W2t, b2, x2, (float*)d_out, nullptr, MROWS, DD, FF_, 0);
}
